// Transformer_4750233829833
// MI455X (gfx1250) — matrix operand order repaired and checked by reading, not compiled, not run on hardware
//
#include <hip/hip_runtime.h>
#include <hip/hip_bf16.h>
#include <math.h>

#define B_    2
#define T_    2048
#define C_    1024
#define H_    16
#define HS_   64
#define NROWS (B_ * T_)   // 4096
#define QKVC  (3 * C_)    // 3072
#define FF4   (4 * C_)    // 4096
#define EPS_  1e-8f

typedef __attribute__((ext_vector_type(16))) _Float16 v16h;
typedef __attribute__((ext_vector_type(8)))  _Float16 v8h;
typedef __attribute__((ext_vector_type(4)))  _Float16 v4h;
typedef __attribute__((ext_vector_type(8)))  float    v8f;

union F16Frag {
  v16h v;
  v8h  h[2];
};

// ---------------------------------------------------------------------------
// f32 -> f16 cast (weights). n must be a multiple of 1024 (true for all uses).
// ---------------------------------------------------------------------------
__global__ __launch_bounds__(256) void cast_f32_f16(const float* __restrict__ in,
                                                    _Float16* __restrict__ out,
                                                    int n) {
  int i = blockIdx.x * 1024 + threadIdx.x * 4;
  if (i + 3 < n) {
    float4 v = *(const float4*)(in + i);
    v4h o = {(_Float16)v.x, (_Float16)v.y, (_Float16)v.z, (_Float16)v.w};
    *(v4h*)(out + i) = o;
  }
}

// ---------------------------------------------------------------------------
// RMSNorm over C_=1024 columns, one block per row, f32 in -> f16 out.
// ---------------------------------------------------------------------------
__global__ __launch_bounds__(256) void rmsnorm_f16(const float* __restrict__ x,
                                                   const float* __restrict__ scale,
                                                   _Float16* __restrict__ out) {
  const int row  = blockIdx.x;
  const int tid  = threadIdx.x;
  const int lane = tid & 31;
  const int wave = tid >> 5;
  const float* xr = x + (long)row * C_;

  float ss = 0.f;
#pragma unroll
  for (int i = tid; i < C_; i += 256) { float v = xr[i]; ss += v * v; }
#pragma unroll
  for (int off = 16; off > 0; off >>= 1) ss += __shfl_xor(ss, off, 32);

  __shared__ float red[8];
  if (lane == 0) red[wave] = ss;
  __syncthreads();
  if (wave == 0) {
    float t = (lane < 8) ? red[lane] : 0.f;
#pragma unroll
    for (int off = 4; off > 0; off >>= 1) t += __shfl_xor(t, off, 32);
    if (lane == 0) red[0] = t;
  }
  __syncthreads();
  const float inv = 1.f / (sqrtf(red[0] * (1.f / (float)C_)) + EPS_);
#pragma unroll
  for (int i = tid; i < C_; i += 256)
    out[(long)row * C_ + i] = (_Float16)(xr[i] * inv * scale[i]);
}

// ---------------------------------------------------------------------------
// Generic WMMA GEMM:  Out[m,n] = sum_k A[m,k]*W[n,k] (+bias[n]) (+resid[m,n])
// A: [M,K] f16 row-major, W: [N,K] f16 row-major (so B-fragments load
// directly from a row-major LDS tile). Block tile 128x128, BK=32,
// 8 waves of 32 lanes, each wave owns a 32x64 subtile (2x4 WMMA tiles).
// This configuration keeps the live set (~230 VGPRs with the backend's
// software pipelining) inside the 256 directly-addressable VGPR window.
// M, N must be multiples of 128; K a multiple of 32.
// ---------------------------------------------------------------------------
template <bool HAS_BIAS, bool HAS_RES, bool OUT_F16>
__global__ __launch_bounds__(256, 1) void gemm_wmma(
    const _Float16* __restrict__ A, const _Float16* __restrict__ W,
    const float* __restrict__ bias, const float* __restrict__ resid,
    void* __restrict__ Out, int K, int ldOut) {
  constexpr int LDT = 40;  // padded f16 row stride (80B, 16B aligned)
  __shared__ __align__(16) _Float16 As[128 * LDT];
  __shared__ __align__(16) _Float16 Ws[128 * LDT];

  const int tid   = threadIdx.x;
  const int lane  = tid & 31;
  const int lr    = lane & 15;   // M (A) / N (B) index within 16
  const int hsel  = lane >> 4;   // lane-half select
  const int wave  = tid >> 5;
  const int waveM = wave & 3;    // 4 waves along M -> 32 rows each
  const int waveN = wave >> 2;   // 2 waves along N -> 64 cols each
  const long m0 = (long)blockIdx.y * 128;
  const long n0 = (long)blockIdx.x * 128;

  const int ldrow  = tid >> 1;        // 0..127
  const int ldhalf = (tid & 1) * 16;  // 0 or 16 (f16 units)

  v8f acc[2][4] = {};

  for (int k0 = 0; k0 < K; k0 += 32) {
    // Stage 128x32 A tile and 128x32 W tile into LDS (32B per thread each).
    const float4* gA = (const float4*)(A + (m0 + ldrow) * K + k0 + ldhalf);
    const float4* gW = (const float4*)(W + (n0 + ldrow) * K + k0 + ldhalf);
    float4 a0 = gA[0], a1 = gA[1];
    float4 w0 = gW[0], w1 = gW[1];
    if (k0 + 32 < K) {  // prefetch next K-slice into caches
      __builtin_prefetch(A + (m0 + ldrow) * K + k0 + 32 + ldhalf, 0, 1);
      __builtin_prefetch(W + (n0 + ldrow) * K + k0 + 32 + ldhalf, 0, 1);
    }
    float4* sA = (float4*)(As + ldrow * LDT + ldhalf);
    float4* sW = (float4*)(Ws + ldrow * LDT + ldhalf);
    sA[0] = a0; sA[1] = a1;
    sW[0] = w0; sW[1] = w1;
    __syncthreads();

    // A fragment: lane lr = M row; lanes 0-15 hold K {0-7,16-23},
    // lanes 16-31 hold K {8-15,24-31}  (ISA 16-bit A 16x32 layout).
    F16Frag af[2];
#pragma unroll
    for (int mi = 0; mi < 2; mi++) {
      const _Float16* p = As + (waveM * 32 + mi * 16 + lr) * LDT + hsel * 8;
      af[mi].h[0] = *(const v8h*)p;
      af[mi].h[1] = *(const v8h*)(p + 16);
    }
    // B fragment: lane lr = N col; lanes 0-15 hold K {0-7,16-23}, lanes 16-31 K {8-15,24-31}.
    F16Frag bf[4];
#pragma unroll
    for (int ni = 0; ni < 4; ni++) {
      const _Float16* p = Ws + (waveN * 64 + ni * 16 + lr) * LDT + hsel * 8;
      bf[ni].h[0] = *(const v8h*)p;
      bf[ni].h[1] = *(const v8h*)(p + 16);
    }
#pragma unroll
    for (int mi = 0; mi < 2; mi++)
#pragma unroll
      for (int ni = 0; ni < 4; ni++)
        acc[mi][ni] = __builtin_amdgcn_wmma_f32_16x16x32_f16(
            false, af[mi].v, false, bf[ni].v, (short)0, acc[mi][ni], false, false);
    __syncthreads();
  }

  // Epilogue: C/D layout -> lane lr = N, VGPR r -> M = r + 8*hsel.
#pragma unroll
  for (int mi = 0; mi < 2; mi++) {
#pragma unroll
    for (int ni = 0; ni < 4; ni++) {
      const long col = n0 + waveN * 64 + ni * 16 + lr;
      const float bv = HAS_BIAS ? bias[col] : 0.f;
#pragma unroll
      for (int r = 0; r < 8; r++) {
        const long row = m0 + waveM * 32 + mi * 16 + r + hsel * 8;
        float v = acc[mi][ni][r] + bv;
        if (HAS_RES) v += resid[row * (long)ldOut + col];
        if (OUT_F16) ((_Float16*)Out)[row * (long)ldOut + col] = (_Float16)v;
        else         ((float*)Out)[row * (long)ldOut + col] = v;
      }
    }
  }
}

// ---------------------------------------------------------------------------
// Fused SwiGLU GEMM:  swi = A @ W1[0:4C].T, gate = A @ W1[4C:8C].T
//   act = relu(swi * sigmoid(swi) * gate)  -> f16 [NROWS, 4C]
// Same tiling as gemm_wmma but with two W tiles / two accumulator banks;
// A-fragments are shared by both banks (16 WMMAs per 20 fragment loads).
// ---------------------------------------------------------------------------
__global__ __launch_bounds__(256, 1) void gemm_swiglu(
    const _Float16* __restrict__ A, const _Float16* __restrict__ W1h,
    _Float16* __restrict__ act) {
  constexpr int LDT = 40;
  __shared__ __align__(16) _Float16 As[128 * LDT];
  __shared__ __align__(16) _Float16 WsS[128 * LDT];
  __shared__ __align__(16) _Float16 WsG[128 * LDT];

  const int tid   = threadIdx.x;
  const int lane  = tid & 31;
  const int lr    = lane & 15;
  const int hsel  = lane >> 4;
  const int wave  = tid >> 5;
  const int waveM = wave & 3;
  const int waveN = wave >> 2;
  const long m0 = (long)blockIdx.y * 128;
  const long n0 = (long)blockIdx.x * 128;

  const int ldrow  = tid >> 1;
  const int ldhalf = (tid & 1) * 16;

  v8f accS[2][4] = {};
  v8f accG[2][4] = {};

  for (int k0 = 0; k0 < C_; k0 += 32) {
    const float4* gA = (const float4*)(A + (m0 + ldrow) * C_ + k0 + ldhalf);
    const float4* gS = (const float4*)(W1h + (n0 + ldrow) * C_ + k0 + ldhalf);
    const float4* gG = (const float4*)(W1h + (n0 + FF4 + ldrow) * C_ + k0 + ldhalf);
    float4 a0 = gA[0], a1 = gA[1];
    float4 s0 = gS[0], s1 = gS[1];
    float4 g0 = gG[0], g1 = gG[1];
    float4* sA = (float4*)(As  + ldrow * LDT + ldhalf);
    float4* sS = (float4*)(WsS + ldrow * LDT + ldhalf);
    float4* sG = (float4*)(WsG + ldrow * LDT + ldhalf);
    sA[0] = a0; sA[1] = a1;
    sS[0] = s0; sS[1] = s1;
    sG[0] = g0; sG[1] = g1;
    __syncthreads();

    F16Frag af[2], bS[4], bG[4];
#pragma unroll
    for (int mi = 0; mi < 2; mi++) {
      const _Float16* p = As + (waveM * 32 + mi * 16 + lr) * LDT + hsel * 8;
      af[mi].h[0] = *(const v8h*)p;
      af[mi].h[1] = *(const v8h*)(p + 16);
    }
#pragma unroll
    for (int ni = 0; ni < 4; ni++) {
      const _Float16* ps = WsS + (waveN * 64 + ni * 16 + lr) * LDT + hsel * 8;
      const _Float16* pg = WsG + (waveN * 64 + ni * 16 + lr) * LDT + hsel * 8;
      bS[ni].h[0] = *(const v8h*)ps; bS[ni].h[1] = *(const v8h*)(ps + 16);
      bG[ni].h[0] = *(const v8h*)pg; bG[ni].h[1] = *(const v8h*)(pg + 16);
    }
#pragma unroll
    for (int mi = 0; mi < 2; mi++)
#pragma unroll
      for (int ni = 0; ni < 4; ni++) {
        accS[mi][ni] = __builtin_amdgcn_wmma_f32_16x16x32_f16(
            false, af[mi].v, false, bS[ni].v, (short)0, accS[mi][ni], false, false);
        accG[mi][ni] = __builtin_amdgcn_wmma_f32_16x16x32_f16(
            false, af[mi].v, false, bG[ni].v, (short)0, accG[mi][ni], false, false);
      }
    __syncthreads();
  }

#pragma unroll
  for (int mi = 0; mi < 2; mi++)
#pragma unroll
    for (int ni = 0; ni < 4; ni++) {
      const long col = n0 + waveN * 64 + ni * 16 + lr;
#pragma unroll
      for (int r = 0; r < 8; r++) {
        const long row = m0 + waveM * 32 + mi * 16 + r + hsel * 8;
        const float swi  = accS[mi][ni][r];
        const float gate = accG[mi][ni][r];
        float sg = swi * (1.f / (1.f + __expf(-swi))) * gate;
        sg = fmaxf(sg, 0.f);
        act[row * (long)FF4 + col] = (_Float16)sg;
      }
    }
}

// ---------------------------------------------------------------------------
// Flash attention: one block = 64 q rows of one (b,h); 4 waves, each wave owns
// 16 q rows. 32-key tiles staged in LDS (V transposed for contiguous PV
// B-fragments). Online softmax via half-wave __shfl_xor row reductions.
// qkv: f16 [B*T, 3C] with head h at cols [h*192, h*192+192) = {q,k,v}.
// av:  f16 [B*T, C]  (b,t,h,hs ordering, matching reference reshape).
// ---------------------------------------------------------------------------
__global__ __launch_bounds__(128, 1) void attn_flash(
    const _Float16* __restrict__ qkv, _Float16* __restrict__ av) {
  __shared__ __align__(16) _Float16 Ks[32 * 64];      // [key][hs]
  __shared__ __align__(16) _Float16 Vt[64 * 32];      // [hs][key] (transposed)
  __shared__ __align__(16) _Float16 Ps[4][16 * 32];   // per-wave P staging

  const int tid  = threadIdx.x;
  const int lane = tid & 31;
  const int lr   = lane & 15;
  const int hsel = lane >> 4;
  const int wave = tid >> 5;
  const int bh = blockIdx.y;
  const int b  = bh >> 4;
  const int h  = bh & 15;
  const int q0 = blockIdx.x * 64 + wave * 16;
  const long rs = QKVC;
  const _Float16* base = qkv + (long)b * T_ * rs + h * (3 * HS_);

  // Q A-fragments for HS chunks [0,32) and [32,64), loaded once from global.
  F16Frag qf[2];
#pragma unroll
  for (int c = 0; c < 2; c++) {
    const _Float16* p = base + (long)(q0 + lr) * rs + c * 32 + hsel * 8;
    qf[c].h[0] = *(const v8h*)p;
    qf[c].h[1] = *(const v8h*)(p + 16);
  }

  float mstate[8], lstate[8];
#pragma unroll
  for (int r = 0; r < 8; r++) { mstate[r] = -__builtin_inff(); lstate[r] = 0.f; }
  v8f accO[4] = {};

  const int krow = tid >> 2;        // 0..31 key row
  const int kq   = (tid & 3) * 16;  // hs chunk of 16

  for (int kt = 0; kt < T_; kt += 32) {
    // K tile: row-major [key][hs].
    const float4* gK = (const float4*)(base + (long)(kt + krow) * rs + HS_ + kq);
    float4 k0v = gK[0], k1v = gK[1];
    *(float4*)(Ks + krow * 64 + kq)     = k0v;
    *(float4*)(Ks + krow * 64 + kq + 8) = k1v;
    // V tile: transpose into [hs][key].
    union { float4 f[2]; _Float16 e[16]; } vv;
    const float4* gV = (const float4*)(base + (long)(kt + krow) * rs + 2 * HS_ + kq);
    vv.f[0] = gV[0]; vv.f[1] = gV[1];
#pragma unroll
    for (int i = 0; i < 16; i++) Vt[(kq + i) * 32 + krow] = vv.e[i];
    __syncthreads();

    // S = Q @ K^T for two 16-key subtiles (2 WMMAs each over HS=64).
    v8f s[2] = {};
#pragma unroll
    for (int kk = 0; kk < 2; kk++)
#pragma unroll
      for (int c = 0; c < 2; c++) {
        F16Frag bf;
        const _Float16* p = Ks + (kk * 16 + lr) * 64 + c * 32 + hsel * 8;
        bf.h[0] = *(const v8h*)p;
        bf.h[1] = *(const v8h*)(p + 16);
        s[kk] = __builtin_amdgcn_wmma_f32_16x16x32_f16(
            false, qf[c].v, false, bf.v, (short)0, s[kk], false, false);
      }

    // Online softmax: row (q) statistics live replicated across 16 lanes.
#pragma unroll
    for (int r = 0; r < 8; r++) {
      float v0 = s[0][r] * 0.125f;   // 1/sqrt(64)
      float v1 = s[1][r] * 0.125f;
      float rm = fmaxf(v0, v1);
#pragma unroll
      for (int off = 1; off < 16; off <<= 1) rm = fmaxf(rm, __shfl_xor(rm, off, 32));
      const float mnew  = fmaxf(mstate[r], rm);
      const float alpha = __expf(mstate[r] - mnew);
      mstate[r] = mnew;
      const float p0 = __expf(v0 - mnew);
      const float p1 = __expf(v1 - mnew);
      s[0][r] = p0; s[1][r] = p1;
      float rsum = p0 + p1;
#pragma unroll
      for (int off = 1; off < 16; off <<= 1) rsum += __shfl_xor(rsum, off, 32);
      lstate[r] = lstate[r] * alpha + rsum;
#pragma unroll
      for (int ni = 0; ni < 4; ni++) accO[ni][r] *= alpha;
    }

    // Re-layout P (C layout) -> A fragment via per-wave LDS staging.
#pragma unroll
    for (int kk = 0; kk < 2; kk++)
#pragma unroll
      for (int r = 0; r < 8; r++)
        Ps[wave][(r + hsel * 8) * 32 + kk * 16 + lr] = (_Float16)s[kk][r];

    F16Frag pf;
    {
      const _Float16* pp = Ps[wave] + lr * 32 + hsel * 8;
      pf.h[0] = *(const v8h*)pp;
      pf.h[1] = *(const v8h*)(pp + 16);
    }
    // O += P @ V : 4 hs-chunks of 16, K=32 keys.
#pragma unroll
    for (int ni = 0; ni < 4; ni++) {
      F16Frag vf;
      const _Float16* vp = Vt + (ni * 16 + lr) * 32 + hsel * 8;
      vf.h[0] = *(const v8h*)vp;
      vf.h[1] = *(const v8h*)(vp + 16);
      accO[ni] = __builtin_amdgcn_wmma_f32_16x16x32_f16(
          false, pf.v, false, vf.v, (short)0, accO[ni], false, false);
    }
    __syncthreads();
  }

  float inv[8];
#pragma unroll
  for (int r = 0; r < 8; r++) inv[r] = 1.f / lstate[r];
#pragma unroll
  for (int ni = 0; ni < 4; ni++)
#pragma unroll
    for (int r = 0; r < 8; r++) {
      const long t = (long)b * T_ + blockIdx.x * 64 + wave * 16 + r + hsel * 8;
      av[t * C_ + h * HS_ + ni * 16 + lr] = (_Float16)(accO[ni][r] * inv[r]);
    }
}

// ---------------------------------------------------------------------------
// Host orchestration.
// ---------------------------------------------------------------------------
extern "C" void kernel_launch(void* const* d_in, const int* in_sizes, int n_in,
                              void* d_out, int out_size, void* d_ws, size_t ws_size,
                              hipStream_t stream) {
  const float* x         = (const float*)d_in[0];
  const float* Wqkv      = (const float*)d_in[1];
  const float* bqkv      = (const float*)d_in[2];
  const float* Wo        = (const float*)d_in[3];
  const float* bo        = (const float*)d_in[4];
  const float* W1        = (const float*)d_in[5];
  const float* W2        = (const float*)d_in[6];
  const float* scale_att = (const float*)d_in[7];
  const float* scale_ff  = (const float*)d_in[8];
  float* out = (float*)d_out;

  char* w = (char*)d_ws;
  auto carve = [&](size_t bytes) -> void* {
    void* p = (void*)w;
    w += (bytes + 255) & ~(size_t)255;
    return p;
  };
  _Float16* h16   = (_Float16*)carve((size_t)NROWS * C_ * 2);
  _Float16* qkv16 = (_Float16*)carve((size_t)NROWS * QKVC * 2);
  _Float16* av16  = (_Float16*)carve((size_t)NROWS * C_ * 2);
  float*    att   = (float*)   carve((size_t)NROWS * C_ * 4);
  _Float16* h216  = (_Float16*)carve((size_t)NROWS * C_ * 2);
  _Float16* act16 = (_Float16*)carve((size_t)NROWS * FF4 * 2);
  _Float16* Wq16  = (_Float16*)carve((size_t)QKVC * C_ * 2);
  _Float16* Wo16  = (_Float16*)carve((size_t)C_ * C_ * 2);
  _Float16* W116  = (_Float16*)carve((size_t)8 * C_ * C_ * 2);
  _Float16* W216  = (_Float16*)carve((size_t)C_ * FF4 * 2);

  cast_f32_f16<<<(QKVC * C_) / 1024, 256, 0, stream>>>(Wqkv, Wq16, QKVC * C_);
  cast_f32_f16<<<(C_ * C_) / 1024, 256, 0, stream>>>(Wo, Wo16, C_ * C_);
  cast_f32_f16<<<(8 * C_ * C_) / 1024, 256, 0, stream>>>(W1, W116, 8 * C_ * C_);
  cast_f32_f16<<<(C_ * FF4) / 1024, 256, 0, stream>>>(W2, W216, C_ * FF4);

  // h = rmsnorm(x) * scale_att
  rmsnorm_f16<<<NROWS, 256, 0, stream>>>(x, scale_att, h16);

  // qkv = h @ Wqkv.T + bqkv   (f16 out)
  gemm_wmma<true, false, true><<<dim3(QKVC / 128, NROWS / 128), 256, 0, stream>>>(
      h16, Wq16, bqkv, nullptr, qkv16, C_, QKVC);

  // attention -> av (f16, [B,T,H,HS] flattened)
  attn_flash<<<dim3(T_ / 64, B_ * H_), 128, 0, stream>>>(qkv16, av16);

  // att_out = av @ Wo.T + bo + x   (f32)
  gemm_wmma<true, true, false><<<dim3(C_ / 128, NROWS / 128), 256, 0, stream>>>(
      av16, Wo16, bo, x, att, C_, C_);

  // h2 = rmsnorm(att_out) * scale_ff
  rmsnorm_f16<<<NROWS, 256, 0, stream>>>(att, scale_ff, h216);

  // act = relu(silu(h2 @ W1a.T) * (h2 @ W1b.T))  (f16)
  gemm_swiglu<<<dim3(FF4 / 128, NROWS / 128), 256, 0, stream>>>(h216, W116, act16);

  // out = act @ W2.T + att_out   (f32)
  gemm_wmma<false, true, false><<<dim3(C_ / 128, NROWS / 128), 256, 0, stream>>>(
      act16, W216, nullptr, att, out, FF4, C_);
}
